// MoEAdapter_73478300500025
// MI455X (gfx1250) — matrix operand order repaired and checked by reading, not compiled, not run on hardware
//
#include <hip/hip_runtime.h>
#include <hip/hip_bf16.h>
#include <math.h>

typedef __attribute__((ext_vector_type(16))) __bf16 v16bf;
typedef __attribute__((ext_vector_type(8)))  float  v8f;

#define DIM    768
#define HID    192
#define NE     4
#define MTILE  64
#define KP     800            // padded K: 768 data + bias rows + zeros -> 25*32
#define KSTEPS 25
#define NT     48             // 768 / 16 N-tiles
#define LDX    808            // LDS row stride in halves (bank-conflict-free, 16B aligned)
#define NFRAG  (KSTEPS * NT)  // 1200 B-fragments per weight matrix

// ---------------------------------------------------------------------------
// Prep: convert w1/w2 (f32) -> bf16, pre-swizzled into WMMA B-fragment order.
// Fragment (kb, nt): 32 lanes x 16 halves contiguous. Lane L: n = nt*16+(L&15),
// holds K = kb*32 + (L>>4)*16 + j  (j = 0..15, packed pairs per VGPR).
// Bias folding: w1s row K==768 = b1 (X gets a ones column);
//               w2s rows 768..771 = b2[e]  (H gets one-hot dispatch columns).
// ---------------------------------------------------------------------------
__global__ void moe_prep(const float* __restrict__ w1, const float* __restrict__ b1,
                         const float* __restrict__ w2, const float* __restrict__ b2,
                         __bf16* __restrict__ w1s, __bf16* __restrict__ w2s) {
    int t = blockIdx.x * blockDim.x + threadIdx.x;
    if (t >= 2 * NFRAG * 32) return;
    int mat  = t / (NFRAG * 32);
    int r    = t % (NFRAG * 32);
    int frag = r >> 5, lane = r & 31;
    int kb   = frag / NT, nt = frag % NT;
    int n    = nt * 16 + (lane & 15);
    int hlf  = lane >> 4;

    union { __bf16 h[16]; v16bf v; } vals;
#pragma unroll
    for (int j = 0; j < 16; ++j) {
        int K = kb * 32 + hlf * 16 + j;
        float val = 0.0f;
        if (mat == 0) {
            if (K < DIM)       val = w1[((size_t)(n / HID) * DIM + K) * HID + (n % HID)];
            else if (K == DIM) val = b1[n];                        // b1 flat [4*192] == concat
        } else {
            if (K < DIM)            val = w2[(size_t)K * DIM + n]; // w2[e][h][n], K==e*192+h
            else if (K < DIM + NE)  val = b2[(size_t)(K - DIM) * DIM + n];
        }
        vals.h[j] = (__bf16)val;
    }
    __bf16* dst = (mat ? w2s : w1s) + ((size_t)frag * 32 + lane) * 16;
    *(v16bf*)dst = vals.v;
}

// ---------------------------------------------------------------------------
// Fused MoE kernel: 1 block = 64 tokens, 256 threads = 8 waves.
// Phase 0: x tile -> LDS bf16 (+ones col). Router in f32 from global, argmax.
// Phase 1: H = mask .* gelu(X @ W1concat)  [64x768] -> LDS (+one-hot cols).
// Phase 2: out = H @ W2concat + x          [64x768] -> global.
// Inner GEMM: double-buffered B frags (hide L2 latency), 4 independent A
// fragment register sets (kill WMMA->ds_load WAR hazard NOPs).
// ---------------------------------------------------------------------------
__global__ __launch_bounds__(256) void moe_main(
        const float* __restrict__ x, const float* __restrict__ rw,
        const float* __restrict__ rb,
        const __bf16* __restrict__ w1s, const __bf16* __restrict__ w2s,
        float* __restrict__ out) {
    extern __shared__ char smem[];
    __bf16* Xs   = (__bf16*)smem;                                  // [64][808]
    __bf16* Hs   = (__bf16*)(smem + (size_t)MTILE * LDX * 2);      // [64][808]
    float*  lg   = (float*)(smem + (size_t)2 * MTILE * LDX * 2);   // [64][4]
    int*    idxs = (int*)((char*)lg + MTILE * NE * 4);             // [64]

    const int tid  = threadIdx.x;
    const int lane = tid & 31;
    const int wave = tid >> 5;
    const int hlf  = lane >> 4;   // 0 | 1
    const int lrow = lane & 15;
    const size_t row0 = (size_t)blockIdx.x * MTILE;

    // -------- phase 0: stage x tile into LDS as bf16 --------
    for (int i = tid; i < MTILE * (DIM / 4); i += 256) {
        int r  = i / (DIM / 4);
        int c4 = i % (DIM / 4);
        const float4 v = *(const float4*)(x + (row0 + r) * DIM + c4 * 4);
        union { __bf16 h[4]; uint2 u; } p;
        p.h[0] = (__bf16)v.x; p.h[1] = (__bf16)v.y;
        p.h[2] = (__bf16)v.z; p.h[3] = (__bf16)v.w;
        *(uint2*)(Xs + (size_t)r * LDX + c4 * 4) = p.u;
    }
    if (tid < MTILE) {
        Xs[(size_t)tid * LDX + DIM] = (__bf16)1.0f;                // bias-fold ones column
        for (int c = DIM + 1; c < KP; ++c) Xs[(size_t)tid * LDX + c] = (__bf16)0.0f;
    }

    // -------- router: f32 logits from global for exact argmax --------
    {
        int token = tid >> 2, e = tid & 3;
        const float* xr = x + (row0 + token) * DIM;
        float acc = rb[e];
        for (int d = 0; d < DIM; ++d) acc = fmaf(xr[d], rw[d * NE + e], acc);
        lg[token * NE + e] = acc;
    }
    __syncthreads();

    if (tid < MTILE) {
        float best = lg[tid * NE]; int bi = 0;
#pragma unroll
        for (int e = 1; e < NE; ++e) {
            float v = lg[tid * NE + e];
            if (v > best) { best = v; bi = e; }                     // first-max, like argmax
        }
        idxs[tid] = bi;
#pragma unroll
        for (int e = 0; e < NE; ++e)                                // one-hot cols -> b2 fold
            Hs[(size_t)tid * LDX + DIM + e] = (__bf16)(e == bi ? 1.0f : 0.0f);
        for (int c = DIM + NE; c < KP; ++c) Hs[(size_t)tid * LDX + c] = (__bf16)0.0f;
    }
    __syncthreads();

    union AF { v16bf v; uint4 u[2]; };

    // A-fragment loads: 4 independent register sets, all issued before use.
#define LOAD_A4(BASE, K0)                                                     \
    {                                                                         \
        const __bf16* ar = (BASE) + (size_t)lrow * LDX + (K0);                \
        a0.u[0] = *(const uint4*)(ar);                                        \
        a0.u[1] = *(const uint4*)(ar + 8);                                    \
        a1.u[0] = *(const uint4*)(ar + 16 * LDX);                             \
        a1.u[1] = *(const uint4*)(ar + 16 * LDX + 8);                         \
        a2.u[0] = *(const uint4*)(ar + 32 * LDX);                             \
        a2.u[1] = *(const uint4*)(ar + 32 * LDX + 8);                         \
        a3.u[0] = *(const uint4*)(ar + 48 * LDX);                             \
        a3.u[1] = *(const uint4*)(ar + 48 * LDX + 8);                         \
    }

#define WMMA4(B)                                                              \
    acc0 = __builtin_amdgcn_wmma_f32_16x16x32_bf16(false, a0.v, false, (B), (short)0, acc0, false, false); \
    acc1 = __builtin_amdgcn_wmma_f32_16x16x32_bf16(false, a1.v, false, (B), (short)0, acc1, false, false); \
    acc2 = __builtin_amdgcn_wmma_f32_16x16x32_bf16(false, a2.v, false, (B), (short)0, acc2, false, false); \
    acc3 = __builtin_amdgcn_wmma_f32_16x16x32_bf16(false, a3.v, false, (B), (short)0, acc3, false, false);

    // -------- phase 1: H = mask .* gelu(X @ W1concat) --------
    for (int s = 0; s < 6; ++s) {
        const int nt = wave * 6 + s;
        const v16bf* bp = (const v16bf*)w1s + (size_t)nt * 32 + lane;
        v8f acc0 = {0,0,0,0,0,0,0,0}, acc1 = acc0, acc2 = acc0, acc3 = acc0;
        v16bf bcur = bp[0];                                        // prefetch kb=0
        for (int kb = 0; kb < KSTEPS - 1; ++kb) {
            AF a0, a1, a2, a3;
            LOAD_A4(Xs, kb * 32 + hlf * 16);
            v16bf bnext = bp[(size_t)(kb + 1) * (NT * 32)];        // prefetch kb+1 (L2)
            WMMA4(bcur);
            bcur = bnext;
        }
        {   // final k-step (no prefetch)
            AF a0, a1, a2, a3;
            LOAD_A4(Xs, (KSTEPS - 1) * 32 + hlf * 16);
            WMMA4(bcur);
        }
        const int e    = nt / (HID / 16);      // which expert this N-strip belongs to
        const int ncol = nt * 16 + lrow;
        v8f* accs[4] = {&acc0, &acc1, &acc2, &acc3};
#pragma unroll
        for (int mt = 0; mt < 4; ++mt) {
#pragma unroll
            for (int v = 0; v < 8; ++v) {
                int m = mt * 16 + v + hlf * 8;
                float h = (*accs[mt])[v];                           // includes b1 via ones col
                float g = 0.5f * h * (1.0f + erff(h * 0.70710678118654752f));
                g = (idxs[m] == e) ? g : 0.0f;                      // top-1 dispatch mask
                Hs[(size_t)m * LDX + ncol] = (__bf16)g;
            }
        }
    }
    __syncthreads();

    // -------- phase 2: out = H @ W2concat + x (b2 via one-hot cols) --------
    for (int s = 0; s < 6; ++s) {
        const int nt = wave * 6 + s;
        const v16bf* bp = (const v16bf*)w2s + (size_t)nt * 32 + lane;
        v8f acc0 = {0,0,0,0,0,0,0,0}, acc1 = acc0, acc2 = acc0, acc3 = acc0;
        v16bf bcur = bp[0];
        for (int kb = 0; kb < KSTEPS - 1; ++kb) {
            AF a0, a1, a2, a3;
            LOAD_A4(Hs, kb * 32 + hlf * 16);
            v16bf bnext = bp[(size_t)(kb + 1) * (NT * 32)];
            WMMA4(bcur);
            bcur = bnext;
        }
        {
            AF a0, a1, a2, a3;
            LOAD_A4(Hs, (KSTEPS - 1) * 32 + hlf * 16);
            WMMA4(bcur);
        }
        const int ncol = nt * 16 + lrow;
        v8f* accs[4] = {&acc0, &acc1, &acc2, &acc3};
#pragma unroll
        for (int mt = 0; mt < 4; ++mt) {
#pragma unroll
            for (int v = 0; v < 8; ++v) {
                int m = mt * 16 + v + hlf * 8;
                size_t gi = (row0 + m) * DIM + ncol;
                out[gi] = (*accs[mt])[v] + x[gi];                   // skip connection (f32)
            }
        }
    }
#undef LOAD_A4
#undef WMMA4
}

// ---------------------------------------------------------------------------
extern "C" void kernel_launch(void* const* d_in, const int* in_sizes, int n_in,
                              void* d_out, int out_size, void* d_ws, size_t ws_size,
                              hipStream_t stream) {
    (void)in_sizes; (void)n_in; (void)out_size; (void)ws_size;
    const float* x  = (const float*)d_in[0];
    const float* rw = (const float*)d_in[1];
    const float* rb = (const float*)d_in[2];
    const float* w1 = (const float*)d_in[3];
    const float* b1 = (const float*)d_in[4];
    const float* w2 = (const float*)d_in[5];
    const float* b2 = (const float*)d_in[6];
    float* out = (float*)d_out;

    __bf16* w1s = (__bf16*)d_ws;                                   // 1.17 MB each, L2-resident
    __bf16* w2s = w1s + (size_t)NFRAG * 32 * 16;

    const int prepThreads = 2 * NFRAG * 32;                        // 76800
    moe_prep<<<(prepThreads + 255) / 256, 256, 0, stream>>>(w1, b1, w2, b2, w1s, w2s);

    const size_t smem = (size_t)2 * MTILE * LDX * 2 + MTILE * NE * 4 + MTILE * 4; // ~203 KB < 320 KB/WGP
    (void)hipFuncSetAttribute((const void*)moe_main,
                              hipFuncAttributeMaxDynamicSharedMemorySize, (int)smem);

    const int nTiles = (16 * 2048) / MTILE;                        // 512 workgroups
    moe_main<<<nTiles, 256, smem, stream>>>(x, rw, rb, w1s, w2s, out);
}
